// LowRankElmanCell_48352741818943
// MI455X (gfx1250) — matrix operand order repaired and checked by reading, not compiled, not run on hardware
//
#include <hip/hip_runtime.h>
#include <hip/hip_bf16.h>
#include <math.h>

// ---------------------------------------------------------------------------
// LowRankElmanCell for MI455X (gfx1250, wave32, WMMA)
// T=1024, B=32, D=1024, R=64
// ---------------------------------------------------------------------------

typedef __attribute__((ext_vector_type(16))) __bf16 v16bf;
typedef __attribute__((ext_vector_type(8)))  float  v8f;

union Frag16 { v16bf bf; unsigned u[8]; };

__device__ __forceinline__ unsigned pack2(float a, float b) {
  union { __bf16 h[2]; unsigned u; } p;
  p.h[0] = (__bf16)a; p.h[1] = (__bf16)b;
  return p.u;
}

// A-operand (16x32 bf16) element position inside a 512-element fragment.
// lanes 0-15 hold K 0-15, lanes 16-31 hold K 16-31 (same k order as B).
__device__ __forceinline__ int a_pos(int mloc, int kin) {
  int half = kin >> 4;
  int i = kin & 15;
  return (mloc + (half << 4)) * 16 + i;
}
// B-operand (32x16 bf16): lanes 0-15 hold K 0-15, lanes 16-31 hold K 16-31.
__device__ __forceinline__ int b_pos(int nloc, int kin) {
  return (nloc + ((kin >> 4) << 4)) * 16 + (kin & 15);
}

// Fragment stored contiguously: lane L owns 32 bytes at frag + L*16 elements.
__device__ __forceinline__ v16bf load_frag(const __bf16* frag, int lane) {
  Frag16 f;
  const unsigned* p = (const unsigned*)frag + lane * 8;
#pragma unroll
  for (int i = 0; i < 8; ++i) f.u[i] = p[i];
  return f.bf;
}

__device__ __forceinline__ v8f wmma_bf16(v16bf a, v16bf b, v8f c) {
  return __builtin_amdgcn_wmma_f32_16x16x32_bf16(false, a, false, b, (short)0, c,
                                                 false, false);
}

// ---------------------------------------------------------------------------
// Kernel 1: 3-step power iteration on UV (in rank space) -> ws[0] = scale
// ---------------------------------------------------------------------------
__device__ void norm1024(float* arr, float* red, int tid, float eps) {
  float p = 0.f;
  for (int d = tid; d < 1024; d += 256) p += arr[d] * arr[d];
  red[tid] = p; __syncthreads();
  for (int s = 128; s > 0; s >>= 1) { if (tid < s) red[tid] += red[tid + s]; __syncthreads(); }
  float inv = 1.f / (sqrtf(red[0]) + eps);
  __syncthreads();
  for (int d = tid; d < 1024; d += 256) arr[d] *= inv;
  __syncthreads();
}

__global__ __launch_bounds__(256) void lre_prep(const float* __restrict__ U,
                                                const float* __restrict__ V,
                                                float* __restrict__ ws) {
  __shared__ float su[1024], sv[1024], st[64], sw[64], red[256];
  int tid = threadIdx.x;
  for (int d = tid; d < 1024; d += 256) {
    float s = sinf((float)(d + 1) * 12.9898f) * 43758.5453f;
    su[d] = s - floorf(s) - 0.5f;   // deterministic pseudo-random start
  }
  __syncthreads();
  norm1024(su, red, tid, 0.f);

  int jj = tid >> 2, cc = tid & 3;
  for (int it = 0; it < 3; ++it) {
    // st = U^T su   (t_j = sum_d U[d,j] u_d)
    { float p = 0.f;
      for (int d = cc * 256; d < cc * 256 + 256; ++d) p += U[d * 64 + jj] * su[d];
      red[tid] = p; }
    __syncthreads();
    if (tid < 64) st[tid] = red[tid*4] + red[tid*4+1] + red[tid*4+2] + red[tid*4+3];
    __syncthreads();
    // sv = V^T st
    for (int d = tid; d < 1024; d += 256) {
      float p = 0.f;
      for (int j = 0; j < 64; ++j) p += V[(size_t)j * 1024 + d] * st[j];
      sv[d] = p;
    }
    __syncthreads();
    norm1024(sv, red, tid, 1e-8f);
    // sw = V sv
    { float p = 0.f;
      for (int d = cc * 256; d < cc * 256 + 256; ++d) p += V[(size_t)jj * 1024 + d] * sv[d];
      red[tid] = p; }
    __syncthreads();
    if (tid < 64) sw[tid] = red[tid*4] + red[tid*4+1] + red[tid*4+2] + red[tid*4+3];
    __syncthreads();
    // su = U sw
    for (int d = tid; d < 1024; d += 256) {
      float p = 0.f;
      for (int j = 0; j < 64; ++j) p += U[d * 64 + j] * sw[j];
      su[d] = p;
    }
    __syncthreads();
    norm1024(su, red, tid, 1e-8f);
  }
  // sigma = (U^T u) . (V v)
  { float p = 0.f;
    for (int d = cc * 256; d < cc * 256 + 256; ++d) p += U[d * 64 + jj] * su[d];
    red[tid] = p; }
  __syncthreads();
  if (tid < 64) st[tid] = red[tid*4] + red[tid*4+1] + red[tid*4+2] + red[tid*4+3];
  __syncthreads();
  { float p = 0.f;
    for (int d = cc * 256; d < cc * 256 + 256; ++d) p += V[(size_t)jj * 1024 + d] * sv[d];
    red[tid] = p; }
  __syncthreads();
  if (tid < 64) sw[tid] = red[tid*4] + red[tid*4+1] + red[tid*4+2] + red[tid*4+3];
  __syncthreads();
  if (tid == 0) {
    float sig = 0.f;
    for (int j = 0; j < 64; ++j) sig += st[j] * sw[j];
    ws[0] = 0.95f / (fabsf(sig) + 1e-8f);
  }
}

// ---------------------------------------------------------------------------
// Kernel 2: XW = x @ W_x^T + b  (32768x1024x1024), bf16 WMMA, f32 accumulate.
// Block = 256 threads (8 waves), 64x64 output tile, K staged in LDS fragments.
// ---------------------------------------------------------------------------
__global__ __launch_bounds__(256) void lre_gemm(const float* __restrict__ X,
                                                const float* __restrict__ Wx,
                                                const float* __restrict__ bias,
                                                float* __restrict__ XW) {
  __shared__ __align__(16) __bf16 sA[4 * 512];
  __shared__ __align__(16) __bf16 sB[4 * 512];
  int tid = threadIdx.x, lane = tid & 31, wave = tid >> 5;
  int rowBase = blockIdx.x * 64;
  int colBase = blockIdx.y * 64;
  int mt = wave & 3, ng = wave >> 2;   // wave -> (M-tile, pair of N-tiles)
  v8f acc0 = {}; v8f acc1 = {};
  for (int kc = 0; kc < 32; ++kc) {
#pragma unroll
    for (int it = 0; it < 4; ++it) {
      int j = tid + it * 256;            // 0..1023 pair jobs for A and for B
      int row = j >> 4, pr = j & 15, kin = pr * 2;
      const float* gx = X + (size_t)(rowBase + row) * 1024 + kc * 32 + kin;
      *(unsigned*)&sA[(row >> 4) * 512 + a_pos(row & 15, kin)] = pack2(gx[0], gx[1]);
      const float* gw = Wx + (size_t)(colBase + row) * 1024 + kc * 32 + kin;
      *(unsigned*)&sB[(row >> 4) * 512 + b_pos(row & 15, kin)] = pack2(gw[0], gw[1]);
    }
    __syncthreads();
    v16bf a  = load_frag(sA + mt * 512, lane);
    v16bf b0 = load_frag(sB + (2 * ng + 0) * 512, lane);
    v16bf b1 = load_frag(sB + (2 * ng + 1) * 512, lane);
    acc0 = wmma_bf16(a, b0, acc0);
    acc1 = wmma_bf16(a, b1, acc1);
    __syncthreads();
  }
  int nl = lane & 15, half = lane >> 4;
#pragma unroll
  for (int rr = 0; rr < 8; ++rr) {
    int m  = rowBase + mt * 16 + rr + 8 * half;
    int e0 = colBase + (2 * ng + 0) * 16 + nl;
    int e1 = colBase + (2 * ng + 1) * 16 + nl;
    XW[(size_t)m * 1024 + e0] = acc0[rr] + bias[e0];
    XW[(size_t)m * 1024 + e1] = acc1[rr] + bias[e1];
  }
}

// ---------------------------------------------------------------------------
// Kernel 3: persistent single-workgroup scan. 512 threads = 16 waves.
// LDS: V-frags 128KB + U-frags 124KB + h-frags 64KB + r-frags 4KB = 320KB.
// The last 2 U e-tiles (4 fragments) live in registers to fit the cap.
// Per step: waves 0-7 compute r = h V^T (256 wmma), all 16 waves compute
// y = r U^T (256 wmma), epilogue: h=tanh(xw+y), out=h*silu(z), store h_all.
// ---------------------------------------------------------------------------
__global__ __launch_bounds__(512) void lre_scan(float* __restrict__ XWO,   // [T][32][1024] in: xW, out: outs
                                                float* __restrict__ HALL,  // [T+1][32][1024]
                                                const float* __restrict__ Z,
                                                const float* __restrict__ Uw,
                                                const float* __restrict__ Vw,
                                                const float* __restrict__ H0,
                                                const float* __restrict__ ws) {
  __shared__ __align__(16) __bf16 sV[128 * 512];  // (jt 0..3)  x (kc 0..31)
  __shared__ __align__(16) __bf16 sU[124 * 512];  // (et 0..61) x (kcj 0..1)
  __shared__ __align__(16) __bf16 sH[ 64 * 512];  // (mt 0..1)  x (kc 0..31)
  __shared__ __align__(16) __bf16 sR[  4 * 512];  // (mt 0..1)  x (kcj 0..1)

  int tid = threadIdx.x, lane = tid & 31, wave = tid >> 5;
  int nl = lane & 15, half = lane >> 4;
  float scale = ws[0];

  // --- build V fragments (scale folded in) ---
  for (int idx = tid; idx < 64 * 512; idx += 512) {
    int j = idx >> 9, d = (idx & 511) << 1;
    unsigned pk = pack2(Vw[(size_t)j * 1024 + d] * scale,
                        Vw[(size_t)j * 1024 + d + 1] * scale);
    *(unsigned*)&sV[(((j >> 4) << 5) + (d >> 5)) * 512 + b_pos(j & 15, d & 31)] = pk;
  }
  // --- build U fragments for e-tiles 0..61 ---
  for (int idx = tid; idx < 992 * 32; idx += 512) {
    int e = idx >> 5, j = (idx & 31) << 1;
    unsigned pk = pack2(Uw[(size_t)e * 64 + j], Uw[(size_t)e * 64 + j + 1]);
    *(unsigned*)&sU[(((e >> 4) << 1) + (j >> 5)) * 512 + b_pos(e & 15, j & 31)] = pk;
  }
  // --- build h fragments from h0 ---
  for (int idx = tid; idx < 32 * 512; idx += 512) {
    int m = idx >> 9, d = (idx & 511) << 1;
    unsigned pk = pack2(H0[(size_t)m * 1024 + d], H0[(size_t)m * 1024 + d + 1]);
    *(unsigned*)&sH[(((m >> 4) << 5) + (d >> 5)) * 512 + a_pos(m & 15, d & 31)] = pk;
  }
  // --- h_all[0] = h0 ---
  for (int idx = tid; idx < 32768; idx += 512) HALL[idx] = H0[idx];

  // --- register-resident U fragments for e-tiles 62,63 (fits LDS in 320KB) ---
  v16bf UB[2][2];
#pragma unroll
  for (int q = 0; q < 2; ++q)
#pragma unroll
    for (int kc = 0; kc < 2; ++kc) {
      Frag16 f;
      int e = (62 + q) * 16 + nl;
#pragma unroll
      for (int d2 = 0; d2 < 8; ++d2) {
        int j = kc * 32 + 2 * d2 + 16 * half;
        f.u[d2] = pack2(Uw[(size_t)e * 64 + j], Uw[(size_t)e * 64 + j + 1]);
      }
      UB[q][kc] = f.bf;
    }
  __syncthreads();

  int mt = wave & 1;
  int ng = wave >> 1;

  for (int t = 0; t < 1024; ++t) {
    // ---------------- phase 1: r = h V^T (waves 0..7, one 16x16 tile each) --
    if (wave < 8) {
      int jt = wave >> 1;
      v8f acc = {};
#pragma unroll 4
      for (int kc = 0; kc < 32; ++kc) {
        v16bf a = load_frag(sH + (mt * 32 + kc) * 512, lane);
        v16bf b = load_frag(sV + (jt * 32 + kc) * 512, lane);
        acc = wmma_bf16(a, b, acc);
      }
#pragma unroll
      for (int rr = 0; rr < 8; ++rr) {
        int mloc = rr + (half << 3);
        int j = jt * 16 + nl;
        sR[((mt << 1) + (j >> 5)) * 512 + a_pos(mloc, j & 31)] = (__bf16)acc[rr];
      }
    }
    __syncthreads();

    // ---------------- phase 2: y = r U^T, nonlinearity, outputs ------------
    v16bf ra0 = load_frag(sR + (mt * 2 + 0) * 512, lane);
    v16bf ra1 = load_frag(sR + (mt * 2 + 1) * 512, lane);
    const float* xwt = XWO + (size_t)t * 32768;
    const float* zt  = Z   + (size_t)t * 32768;
    float* ot  = XWO + (size_t)t * 32768;
    float* ht1 = HALL + (size_t)(t + 1) * 32768;
#pragma unroll
    for (int q = 0; q < 8; ++q) {
      int nt = ng * 8 + q;
      v16bf b0, b1;
      if (nt >= 62) { b0 = UB[nt - 62][0]; b1 = UB[nt - 62][1]; }
      else {
        b0 = load_frag(sU + (nt * 2 + 0) * 512, lane);
        b1 = load_frag(sU + (nt * 2 + 1) * 512, lane);
      }
      v8f acc = {};
      acc = wmma_bf16(ra0, b0, acc);
      acc = wmma_bf16(ra1, b1, acc);
#pragma unroll
      for (int rr = 0; rr < 8; ++rr) {
        int mloc = rr + (half << 3);
        int m = mt * 16 + mloc;
        int e = nt * 16 + nl;
        size_t off = (size_t)m * 1024 + e;
        float hnew = tanhf(xwt[off] + acc[rr]);
        sH[((mt << 5) + (e >> 5)) * 512 + a_pos(mloc, e & 31)] = (__bf16)hnew;
        ht1[off] = hnew;
        float zz = zt[off];
        ot[off] = hnew * (zz / (1.f + __expf(-zz)));   // h * silu(z)
      }
    }
    // prefetch next step's xw / z slabs (128KB each) into cache
    if (t < 1023) {
      __builtin_prefetch((const char*)(xwt + 32768) + tid * 256, 0, 1);
      __builtin_prefetch((const char*)(zt  + 32768) + tid * 256, 0, 1);
    }
    __syncthreads();
  }
}

// ---------------------------------------------------------------------------
extern "C" void kernel_launch(void* const* d_in, const int* in_sizes, int n_in,
                              void* d_out, int out_size, void* d_ws, size_t ws_size,
                              hipStream_t stream) {
  const float* x  = (const float*)d_in[0];
  const float* z  = (const float*)d_in[1];
  const float* h0 = (const float*)d_in[2];
  const float* Wx = (const float*)d_in[3];
  const float* U  = (const float*)d_in[4];
  const float* V  = (const float*)d_in[5];
  const float* b  = (const float*)d_in[6];
  float* out  = (float*)d_out;
  float* xw   = out;                               // outs region, reused as xW staging
  float* hall = out + (size_t)1024 * 32 * 1024;    // h_all region
  float* ws   = (float*)d_ws;

  lre_prep<<<1, 256, 0, stream>>>(U, V, ws);
  lre_gemm<<<dim3(512, 16), 256, 0, stream>>>(x, Wx, b, xw);
  lre_scan<<<1, 512, 0, stream>>>(xw, hall, z, U, V, h0, ws);
}
